// EventGNN_53961969107164
// MI455X (gfx1250) — compile-verified, matrix operand order checked by reading, not run on hardware
//
#include <hip/hip_runtime.h>

typedef __attribute__((ext_vector_type(16))) __bf16 v16bf;
typedef __attribute__((ext_vector_type(8)))  float  v8f;

#define NNODES 50000
#define NEDGES 500000
#define NGRAPH 64
#define BM 128
#define EPSV 1e-5f

// fragment interleave (ISA 7.12.2, 16-bit operands, 16x16x32):
//   k = j + 8*(half + (j>=8))  <=>  half=(k>>3)&1 ; j=(k&7)+8*(k>>4)

// CDNA5 async global->LDS copy (16B per lane), tracked by ASYNCcnt.
__device__ __forceinline__ void async_copy16(const void* gsrc, void* ldst) {
    unsigned lds = (unsigned)(uintptr_t)ldst;          // LDS aperture: addr[31:0]
    asm volatile("global_load_async_to_lds_b128 %0, %1, off"
                 :: "v"(lds), "v"(gsrc) : "memory");
}
__device__ __forceinline__ void async_wait0() {
    asm volatile("s_wait_asynccnt 0x0" ::: "memory");
}

// ---------------------------------------------------------------- utilities
__global__ void zero_f32(float* p, int n) {
    int i = blockIdx.x * blockDim.x + threadIdx.x;
    if (i < n) p[i] = 0.f;
}

// f32 weight [K,OC] -> bf16 in WMMA fragment order [KT][OC/16][32][16],
// zero-padding K..KPAD
__global__ void wconv_kernel(const float* __restrict__ W, __bf16* __restrict__ Wf,
                             int K, int KPAD, int OC) {
    int i = blockIdx.x * blockDim.x + threadIdx.x;
    if (i >= KPAD * OC) return;
    int j   = i & 15;
    int ln  = (i >> 4) & 31;
    int t   = i >> 9;                       // kt*(OC/16) + ct
    int NCT = OC >> 4;
    int kt  = t / NCT, ct = t - kt * NCT;
    int hf  = ln >> 4, n = ln & 15;
    int k   = kt * 32 + j + 8 * (hf + (j >= 8 ? 1 : 0));
    int c   = ct * 16 + n;
    Wf[i] = (__bf16)((k < K) ? W[k * OC + c] : 0.f);
}

// per-channel batch stats -> affine (scale, shift);  BN(h) = h*scale + shift
__global__ void stats_kernel(const float* __restrict__ sum, const float* __restrict__ sumsq,
                             const float* __restrict__ g, const float* __restrict__ beta,
                             float invE, int OC,
                             float* __restrict__ scale, float* __restrict__ shift) {
    int c = blockIdx.x * blockDim.x + threadIdx.x;
    if (c >= OC) return;
    float m = sum[c] * invE;
    float v = sumsq[c] * invE - m * m;
    float s = g[c] * rsqrtf(v + EPSV);
    scale[c] = s;
    shift[c] = beta[c] - m * s;
}

__global__ void count_kernel(const int* __restrict__ dst, float* __restrict__ cnt, int E) {
    int e = blockIdx.x * blockDim.x + threadIdx.x;
    if (e < E) atomicAdd(&cnt[dst[e]], 1.f);
}

// ---------------------------------------------------------- fused edge GEMM
// MODE 0: A tile = gather/concat [x_dst , x_src - x_dst] from node features (C = K/2)
// MODE 1: A tile = relu(h_in*scale + shift)   (h_in is [E,K] f32, K==KPAD)
// A is staged into LDS in fragment order; the whole fragment-ordered W matrix
// is async-copied into LDS once (global_load_async_to_lds_b128 + ASYNCcnt).
template <int OC, int MODE>
__global__ __launch_bounds__(256)
void edge_gemm(const float* __restrict__ nf,
               const int* __restrict__ srcIdx, const int* __restrict__ dstIdx,
               const float* __restrict__ hin,
               const float* __restrict__ scale, const float* __restrict__ shift,
               const __bf16* __restrict__ Wf,
               int E, int K, int KPAD,
               float* __restrict__ hout,
               float* __restrict__ chsum, float* __restrict__ chsumsq) {
    extern __shared__ __align__(32) char smem[];
    __bf16* sA = (__bf16*)smem;            // frag order: [8 rowtiles][KT][32 lanes][16]
    __bf16* sW = sA + BM * KPAD;           // frag order: [KT][OC/16][32 lanes][16]

    const int tid  = threadIdx.x;
    const int wave = tid >> 5;
    const int lane = tid & 31;
    const int e0   = blockIdx.x * BM;
    const int C    = K >> 1;
    const int KT   = KPAD >> 5;            // number of 32-wide K tiles
    const int NCT  = OC >> 4;              // number of 16-wide col tiles

    // ---- async-stage whole fragment-ordered W into LDS (contiguous copy)
    {
        const char* gw = (const char*)Wf;
        char* lw = (char*)sW;
        int chunks = (KPAD * OC) >> 3;     // 16B chunks
        for (int f = tid; f < chunks; f += 256)
            async_copy16(gw + ((size_t)f << 4), lw + ((size_t)f << 4));
    }

    // ---- stage A tile into LDS fragment order
    if (MODE == 0) {
        for (int f = tid; f < BM * KPAD; f += 256) {
            int r = f / KPAD, c = f - r * KPAD;
            int e = e0 + r;
            float v = 0.f;
            if (e < E && c < K) {
                int di = dstIdx[e];
                if (c < C) {
                    v = nf[(long)di * C + c];
                } else {
                    int si = srcIdx[e];
                    int cc = c - C;
                    v = nf[(long)si * C + cc] - nf[(long)di * C + cc];
                }
            }
            int rt = r >> 4, m = r & 15;
            int kt = c >> 5, kk = c & 31;
            int hf = (kk >> 3) & 1, j = (kk & 7) + ((kk >> 4) << 3);
            int ln = m + (hf << 4);
            sA[(((rt * KT + kt) << 5) + ln) * 16 + j] = (__bf16)v;
        }
    } else {
        const int KQ = KPAD >> 2;          // float4 quads per row
        for (int q = tid; q < BM * KQ; q += 256) {
            int r = q / KQ, c = (q - r * KQ) << 2;
            int e = e0 + r;
            float4 t4 = make_float4(0.f, 0.f, 0.f, 0.f);
            if (e < E) t4 = ((const float4*)hin)[((long)e * KPAD + c) >> 2];
            float vv[4] = {t4.x, t4.y, t4.z, t4.w};
            int rt = r >> 4, m = r & 15;
            int kt = c >> 5, kk = c & 31;
            int hf = (kk >> 3) & 1, j0 = (kk & 7) + ((kk >> 4) << 3);
            int ln = m + (hf << 4);
            __bf16* dstp = &sA[(((rt * KT + kt) << 5) + ln) * 16 + j0];
#pragma unroll
            for (int u = 0; u < 4; ++u) {
                float t = vv[u] * scale[c + u] + shift[c + u];
                dstp[u] = (__bf16)(t > 0.f ? t : 0.f);
            }
        }
    }

    async_wait0();                         // our wave's W transfers complete
    __syncthreads();                       // everyone's A stores + W transfers visible

    const v16bf* sAf = (const v16bf*)sA;
    const v16bf* sWf = (const v16bf*)sW;

    v8f acc[OC / 16];
#pragma unroll
    for (int t = 0; t < OC / 16; ++t) {
        v8f z = {0.f, 0.f, 0.f, 0.f, 0.f, 0.f, 0.f, 0.f};
        acc[t] = z;
    }

    // ---- barrier-free K loop: pure ds_load_b128 + v_wmma
    for (int kt = 0; kt < KT; ++kt) {
        v16bf a = sAf[(wave * KT + kt) * 32 + lane];       // 2x ds_load_b128
#pragma unroll
        for (int ct = 0; ct < OC / 16; ++ct) {
            v16bf b = sWf[((kt * NCT + ct) << 5) + lane];  // 2x ds_load_b128
            acc[ct] = __builtin_amdgcn_wmma_f32_16x16x32_bf16(
                false, a, false, b, (short)0, acc[ct], false, false);
        }
    }

    // ---- epilogue: store f32; reduce BN stats through LDS, then 2*OC global atomics
    __syncthreads();                       // done reading sA/sW: reuse as reduction scratch
    float* redsum = (float*)smem;          // [OC]
    float* redsq  = redsum + OC;           // [OC]
    for (int i = tid; i < 2 * OC; i += 256) redsum[i] = 0.f;
    __syncthreads();

    const int m = lane & 15, hlf = lane >> 4;
#pragma unroll
    for (int ct = 0; ct < OC / 16; ++ct) {
        int col = ct * 16 + m;
        float s = 0.f, sq = 0.f;
#pragma unroll
        for (int r = 0; r < 8; ++r) {
            int e = e0 + wave * 16 + hlf * 8 + r;          // C/D layout: M = r + 8*half
            float val = acc[ct][r];
            if (e < E) {
                hout[(long)e * OC + col] = val;
                s  += val;
                sq += val * val;
            }
        }
        atomicAdd(&redsum[col], s);
        atomicAdd(&redsq[col], sq);
    }
    __syncthreads();
    for (int i = tid; i < OC; i += 256) {
        atomicAdd(&chsum[i],   redsum[i]);
        atomicAdd(&chsumsq[i], redsq[i]);
    }
}

// --------------------------------------------- BN+ReLU + scatter-mean (sum)
__global__ void aggregate_kernel(const float* __restrict__ h,
                                 const float* __restrict__ scale, const float* __restrict__ shift,
                                 const int* __restrict__ dst,
                                 float* __restrict__ acc, int E, int OC) {
    const int OQ = OC >> 2;
    long i = (long)blockIdx.x * blockDim.x + threadIdx.x;
    if (i >= (long)E * OQ) return;
    int e = (int)(i / OQ), c = (int)(i - (long)e * OQ) << 2;
    float4 v = ((const float4*)h)[i];
    float vv[4] = {v.x, v.y, v.z, v.w};
    float* ap = &acc[(long)dst[e] * OC + c];
#pragma unroll
    for (int u = 0; u < 4; ++u) {
        float t = vv[u] * scale[c + u] + shift[c + u];
        atomicAdd(&ap[u], t > 0.f ? t : 0.f);
    }
}

__global__ void divide_kernel(const float* __restrict__ acc, const float* __restrict__ cnt,
                              float* __restrict__ nfout, int N, int OC) {
    const int OQ = OC >> 2;
    long i = (long)blockIdx.x * blockDim.x + threadIdx.x;
    if (i >= (long)N * OQ) return;
    int n = (int)(i / OQ);
    float inv = 1.f / fmaxf(cnt[n], 1.f);
    float4 v = ((const float4*)acc)[i];
    v.x *= inv; v.y *= inv; v.z *= inv; v.w *= inv;
    ((float4*)nfout)[i] = v;
}

// -------------------------------------------------------------- mean pool
__global__ void gcnt_kernel(const int* __restrict__ batch, float* __restrict__ gcnt, int N) {
    int n = blockIdx.x * blockDim.x + threadIdx.x;
    if (n < N) atomicAdd(&gcnt[batch[n]], 1.f);
}

__global__ void gsum_kernel(const float* __restrict__ nf, const int* __restrict__ batch,
                            float* __restrict__ gsum, int N, int OC) {
    const int OQ = OC >> 2;
    long i = (long)blockIdx.x * blockDim.x + threadIdx.x;
    if (i >= (long)N * OQ) return;
    int n = (int)(i / OQ), c = (int)(i - (long)n * OQ) << 2;
    float4 v = ((const float4*)nf)[i];
    float vv[4] = {v.x, v.y, v.z, v.w};
    float* gp = &gsum[(long)batch[n] * OC + c];
#pragma unroll
    for (int u = 0; u < 4; ++u) atomicAdd(&gp[u], vv[u]);
}

// -------------------------------------------------------------- MLP head
__global__ __launch_bounds__(256)
void head_kernel(const float* __restrict__ gsum, const float* __restrict__ gcnt,
                 const float* __restrict__ W1, const float* __restrict__ b1,
                 const float* __restrict__ W2, const float* __restrict__ b2,
                 float* __restrict__ out) {
    __shared__ float z[NGRAPH * 128];
    int tid = threadIdx.x;
    for (int i = tid; i < NGRAPH * 128; i += 256) {
        int g = i / 128, j = i - g * 128;
        float a = b1[j];
        float inv = 1.f / fmaxf(gcnt[g], 1.f);
        for (int k = 0; k < 256; ++k)
            a += (gsum[g * 256 + k] * inv) * W1[k * 128 + j];
        z[i] = a > 0.f ? a : 0.f;
    }
    __syncthreads();
    for (int i = tid; i < NGRAPH * 2; i += 256) {
        int g = i / 2, o = i - g * 2;
        float a = b2[o];
        for (int j = 0; j < 128; ++j) a += z[g * 128 + j] * W2[j * 2 + o];
        out[i] = a;
    }
}

// ================================================================= driver
extern "C" void kernel_launch(void* const* d_in, const int* in_sizes, int n_in,
                              void* d_out, int out_size, void* d_ws, size_t ws_size,
                              hipStream_t stream) {
    const int N = NNODES, E = NEDGES;
    const float* x      = (const float*)d_in[0];
    const int*   eidx   = (const int*)d_in[1];
    const int*   batch  = (const int*)d_in[2];
    const int*   src    = eidx;
    const int*   dstp   = eidx + E;
    const float* headW1 = (const float*)d_in[27];
    const float* headB1 = (const float*)d_in[28];
    const float* headW2 = (const float*)d_in[29];
    const float* headB2 = (const float*)d_in[30];

    // workspace carve-out
    char* ws = (char*)d_ws;
    size_t off = 0;
    auto carve = [&](size_t bytes) { size_t o = off; off += (bytes + 255) & ~(size_t)255; return o; };
    float*  h      = (float*)(ws + carve((size_t)E * 256 * 4));   // activation buffer (in-place reuse)
    float*  nfA    = (float*)(ws + carve((size_t)N * 256 * 4));
    float*  nfB    = (float*)(ws + carve((size_t)N * 256 * 4));
    float*  acc    = (float*)(ws + carve((size_t)N * 256 * 4));
    float*  cnt    = (float*)(ws + carve((size_t)N * 4));
    __bf16* Wbf    = (__bf16*)(ws + carve((size_t)256 * 256 * 2));
    float*  sum1   = (float*)(ws + carve(256 * 4));
    float*  sumsq1 = (float*)(ws + carve(256 * 4));
    float*  scl1   = (float*)(ws + carve(256 * 4));
    float*  shf1   = (float*)(ws + carve(256 * 4));
    float*  sum2   = (float*)(ws + carve(256 * 4));
    float*  sumsq2 = (float*)(ws + carve(256 * 4));
    float*  scl2   = (float*)(ws + carve(256 * 4));
    float*  shf2   = (float*)(ws + carve(256 * 4));
    float*  gsum   = (float*)(ws + carve((size_t)NGRAPH * 256 * 4));
    float*  gcnt   = (float*)(ws + carve((size_t)NGRAPH * 4));

    const float invE = 1.f / (float)E;
    const int gemmGrid = (E + BM - 1) / BM;

    // node in-degree counts (shared by all three blocks)
    zero_f32<<<(N + 255) / 256, 256, 0, stream>>>(cnt, N);
    count_kernel<<<(E + 255) / 256, 256, 0, stream>>>(dstp, cnt, E);

    struct Blk { int C, K, KPAD, OC; const float *W1, *g1, *be1, *W2, *g2, *be2; const float *in; float *out; };
    Blk blks[3] = {
        {  7,  14,  32,  64, (const float*)d_in[3],  (const float*)d_in[5],  (const float*)d_in[6],
                             (const float*)d_in[7],  (const float*)d_in[9],  (const float*)d_in[10], x,   nfA },
        { 64, 128, 128, 128, (const float*)d_in[11], (const float*)d_in[13], (const float*)d_in[14],
                             (const float*)d_in[15], (const float*)d_in[17], (const float*)d_in[18], nfA, nfB },
        {128, 256, 256, 256, (const float*)d_in[19], (const float*)d_in[21], (const float*)d_in[22],
                             (const float*)d_in[23], (const float*)d_in[25], (const float*)d_in[26], nfB, nfA },
    };

    for (int b = 0; b < 3; ++b) {
        const Blk& B = blks[b];
        const int OC = B.OC;
        zero_f32<<<(OC + 255) / 256, 256, 0, stream>>>(sum1, OC);
        zero_f32<<<(OC + 255) / 256, 256, 0, stream>>>(sumsq1, OC);
        zero_f32<<<(OC + 255) / 256, 256, 0, stream>>>(sum2, OC);
        zero_f32<<<(OC + 255) / 256, 256, 0, stream>>>(sumsq2, OC);
        zero_f32<<<(int)(((long)N * OC + 255) / 256), 256, 0, stream>>>(acc, N * OC);

        // ---- linear 1 (fused gather/concat) + stats
        wconv_kernel<<<(B.KPAD * OC + 255) / 256, 256, 0, stream>>>(B.W1, Wbf, B.K, B.KPAD, OC);
        size_t sh1 = (size_t)BM * B.KPAD * 2 + (size_t)B.KPAD * OC * 2;
        if (OC == 64)
            edge_gemm<64, 0><<<gemmGrid, 256, sh1, stream>>>(B.in, src, dstp, nullptr, nullptr, nullptr,
                                                             Wbf, E, B.K, B.KPAD, h, sum1, sumsq1);
        else if (OC == 128)
            edge_gemm<128, 0><<<gemmGrid, 256, sh1, stream>>>(B.in, src, dstp, nullptr, nullptr, nullptr,
                                                              Wbf, E, B.K, B.KPAD, h, sum1, sumsq1);
        else
            edge_gemm<256, 0><<<gemmGrid, 256, sh1, stream>>>(B.in, src, dstp, nullptr, nullptr, nullptr,
                                                              Wbf, E, B.K, B.KPAD, h, sum1, sumsq1);
        stats_kernel<<<(OC + 63) / 64, 64, 0, stream>>>(sum1, sumsq1, B.g1, B.be1, invE, OC, scl1, shf1);

        // ---- linear 2 (BN1+ReLU fused into A-staging, in-place h) + stats
        wconv_kernel<<<(OC * OC + 255) / 256, 256, 0, stream>>>(B.W2, Wbf, OC, OC, OC);
        size_t sh2 = (size_t)BM * OC * 2 + (size_t)OC * OC * 2;
        if (OC == 64)
            edge_gemm<64, 1><<<gemmGrid, 256, sh2, stream>>>(nullptr, src, dstp, h, scl1, shf1,
                                                             Wbf, E, OC, OC, h, sum2, sumsq2);
        else if (OC == 128)
            edge_gemm<128, 1><<<gemmGrid, 256, sh2, stream>>>(nullptr, src, dstp, h, scl1, shf1,
                                                              Wbf, E, OC, OC, h, sum2, sumsq2);
        else
            edge_gemm<256, 1><<<gemmGrid, 256, sh2, stream>>>(nullptr, src, dstp, h, scl1, shf1,
                                                              Wbf, E, OC, OC, h, sum2, sumsq2);
        stats_kernel<<<(OC + 63) / 64, 64, 0, stream>>>(sum2, sumsq2, B.g2, B.be2, invE, OC, scl2, shf2);

        // ---- BN2+ReLU + scatter-mean into nodes
        long totq = (long)E * (OC >> 2);
        aggregate_kernel<<<(int)((totq + 255) / 256), 256, 0, stream>>>(h, scl2, shf2, dstp, acc, E, OC);
        divide_kernel<<<(int)(((long)N * (OC >> 2) + 255) / 256), 256, 0, stream>>>(acc, cnt, B.out, N, OC);
    }

    // ---- global mean pool + head
    zero_f32<<<(NGRAPH * 256 + 255) / 256, 256, 0, stream>>>(gsum, NGRAPH * 256);
    zero_f32<<<1, 64, 0, stream>>>(gcnt, NGRAPH);
    gcnt_kernel<<<(N + 255) / 256, 256, 0, stream>>>(batch, gcnt, N);
    gsum_kernel<<<(int)(((long)N * 64 + 255) / 256), 256, 0, stream>>>(nfA, batch, gsum, N, 256);
    head_kernel<<<1, 256, 0, stream>>>(gsum, gcnt, headW1, headB1, headW2, headB2, (float*)d_out);
}
